// Forecaster_PONI_39573828665789
// MI455X (gfx1250) — matrix operand order repaired and checked by reading, not compiled, not run on hardware
//
#include <hip/hip_runtime.h>
#include <math.h>

// ---------------------------------------------------------------------------
// CDNA5 (gfx1250, wave32) implicit-GEMM convolution via V_WMMA_F32_16X16X32_F16
// Block = 128 threads = 4 wave32. Block tile: 32 cout x 64 positions.
// Each wave owns one 16-row cout sub-tile and two 16-col position sub-tiles
// (2 WMMAs per K-chunk sharing the A fragment). K staged in chunks of 32.
// ---------------------------------------------------------------------------
typedef __attribute__((ext_vector_type(16))) _Float16 v16h;
typedef __attribute__((ext_vector_type(8)))  _Float16 v8h;
typedef __attribute__((ext_vector_type(8)))  float    v8f;

#define LDSTRIDE 40   // padded LDS row stride in halves (80B: 16B-aligned rows)

// MODE 0: 3x3 conv, pad = (stride==2 ? 0 : 1)  [TF-style SAME]
// MODE 1: 4x4 stride-2 transposed conv, pad_lo = 2 (gather over dilated input)
// act: 0 = none, 1 = leaky-relu(0.2), 2 = sigmoid.  Input = concat(x1[C1], x2[C2]).
template <int MODE>
__launch_bounds__(128)
__global__ void wmma_conv_kernel(const float* __restrict__ x1, int C1,
                                 const float* __restrict__ x2, int C2,
                                 const float* __restrict__ wgt,
                                 const float* __restrict__ bias,
                                 float* __restrict__ out,
                                 int N, int Hin, int Win, int Hout, int Wout,
                                 int Cout, int stride, int act)
{
    constexpr int KW_ = (MODE == 0) ? 3 : 4;
    constexpr int KHW = (MODE == 0) ? 9 : 16;

    __shared__ __align__(16) _Float16 Ws[32 * LDSTRIDE]; // [cout_local][k_local]
    __shared__ __align__(16) _Float16 Xs[64 * LDSTRIDE]; // [pos_local][k_local]

    const int pad = (MODE == 0) ? ((stride == 2) ? 0 : 1) : 2;
    const int Cin = C1 + C2;
    const int K   = Cin * KHW;
    const int HWo = Hout * Wout;
    const int totalPos = N * HWo;

    const int posBase = blockIdx.x * 64;
    const int coBase  = blockIdx.y * 32;
    const int tid   = threadIdx.x;
    const int lane  = tid & 31;
    const int col   = tid & 31;     // fixed K-chunk column staged by this thread
    const int rbase = tid >> 5;     // row base (rows rbase+4j)
    const int wy = (tid >> 6) & 1;  // wave cout sub-tile
    const int wx = (tid >> 5) & 1;  // wave position sub-tile pair

    // ---- hoist position decomposition (loop-invariant over K) ----
    int pn[16], py[16], px[16];
#pragma unroll
    for (int j = 0; j < 16; ++j) {
        int pos = posBase + rbase + 4 * j;
        if (pos < totalPos) {
            int n  = pos / HWo;
            int rp = pos - n * HWo;
            int oy = rp / Wout;
            pn[j] = n; py[j] = oy; px[j] = rp - oy * Wout;
        } else {
            pn[j] = -1; py[j] = 0; px[j] = 0;
        }
    }

    v8f acc0 = {};
    v8f acc1 = {};

    for (int kc = 0; kc < K; kc += 32) {
        // one k-decomposition per thread per chunk (column is fixed)
        const int  k  = kc + col;
        const bool kv = (k < K);
        const int  kk = kv ? k : 0;
        const int  ci = kk / KHW;            // compile-time divisor (9 or 16)
        const int  rk = kk - ci * KHW;
        const int  ky = rk / KW_;
        const int  kx = rk - ky * KW_;
        const bool useX2 = (ci >= C1);
        const float* __restrict__ src = useX2 ? x2 : x1;
        const int  cc = useX2 ? (ci - C1) : ci;
        const int  Cs = useX2 ? C2 : C1;
        const int  cbase = (cc * Hin) * Win; // channel/batch-invariant part added below

        // ---- weights: 8 rows, branchless clamped-address loads ----
        float wv[8];
#pragma unroll
        for (int j = 0; j < 8; ++j) {
            int  co  = coBase + rbase + 4 * j;
            bool v   = kv && (co < Cout);
            int  idx = v ? (co * K + kk) : 0;
            float w  = wgt[idx];
            wv[j] = v ? w : 0.0f;
        }

        // ---- im2col gather: 16 rows, branchless clamped-address loads ----
        float xv[16];
#pragma unroll
        for (int j = 0; j < 16; ++j) {
            bool v = kv && (pn[j] >= 0);
            int iy, ix;
            if (MODE == 0) {
                iy = py[j] * stride + ky - pad;
                ix = px[j] * stride + kx - pad;
                v = v && (iy >= 0) && (iy < Hin) && (ix >= 0) && (ix < Win);
            } else {
                int ty = py[j] + ky - pad;
                int tx = px[j] + kx - pad;
                iy = ty >> 1; ix = tx >> 1;
                v = v && (ty >= 0) && (tx >= 0) && ((ty & 1) == 0) && ((tx & 1) == 0)
                      && (iy < Hin) && (ix < Win);
            }
            int idx = v ? ((pn[j] * Cs * Hin + iy) * Win + ix + cbase) : 0;
            float xr = src[idx];
            xv[j] = v ? xr : 0.0f;
        }

        // ---- convert + stage to LDS ----
#pragma unroll
        for (int j = 0; j < 8; ++j)
            Ws[(rbase + 4 * j) * LDSTRIDE + col] = (_Float16)wv[j];
#pragma unroll
        for (int j = 0; j < 16; ++j)
            Xs[(rbase + 4 * j) * LDSTRIDE + col] = (_Float16)xv[j];
        __syncthreads();

        // ---- A fragment (16x32 f16): lanes<16 -> K 0-15; lanes>=16 -> K 16-31
        {
            int mrow  = wy * 16 + (lane & 15);
            int abase = (lane < 16) ? 0 : 16;
            v8h a0 = *(const v8h*)&Ws[mrow * LDSTRIDE + abase];
            v8h a1 = *(const v8h*)&Ws[mrow * LDSTRIDE + abase + 8];
            v16h A = __builtin_shufflevector(a0, a1, 0,1,2,3,4,5,6,7,8,9,10,11,12,13,14,15);

            // ---- B fragments (32x16): lanes<16 -> K 0-15; lanes>=16 -> K 16-31
            int nc    = wx * 16 + (lane & 15);
            int bbase = (lane < 16) ? 0 : 16;
            v8h b0 = *(const v8h*)&Xs[nc * LDSTRIDE + bbase];
            v8h b1 = *(const v8h*)&Xs[nc * LDSTRIDE + bbase + 8];
            v16h B0 = __builtin_shufflevector(b0, b1, 0,1,2,3,4,5,6,7,8,9,10,11,12,13,14,15);
            v8h c0 = *(const v8h*)&Xs[(nc + 32) * LDSTRIDE + bbase];
            v8h c1 = *(const v8h*)&Xs[(nc + 32) * LDSTRIDE + bbase + 8];
            v16h B1 = __builtin_shufflevector(c0, c1, 0,1,2,3,4,5,6,7,8,9,10,11,12,13,14,15);

            acc0 = __builtin_amdgcn_wmma_f32_16x16x32_f16(false, A, false, B0,
                                                          (short)0, acc0, false, false);
            acc1 = __builtin_amdgcn_wmma_f32_16x16x32_f16(false, A, false, B1,
                                                          (short)0, acc1, false, false);
        }
        __syncthreads();
    }

    // ---- epilogue: bias + activation, scatter to NCHW (two position tiles) ----
    const int cob = coBase + wy * 16 + ((lane < 16) ? 0 : 8);
#pragma unroll
    for (int half = 0; half < 2; ++half) {
        int pos = posBase + half * 32 + wx * 16 + (lane & 15);
        if (pos < totalPos) {
            int n  = pos / HWo;
            int rp = pos - n * HWo;
            int oy = rp / Wout;
            int ox = rp - oy * Wout;
            const v8f& acc = half ? acc1 : acc0;
#pragma unroll
            for (int i = 0; i < 8; ++i) {
                int co = cob + i;
                if (co < Cout) {
                    float v = acc[i] + bias[co];
                    if (act == 1)      v = (v > 0.0f) ? v : 0.2f * v;
                    else if (act == 2) v = 1.0f / (1.0f + __expf(-v));
                    out[((size_t)(n * Cout + co) * Hout + oy) * Wout + ox] = v;
                }
            }
        }
    }
}

// ---------------------------------------------------------------------------
// Elementwise glue
// ---------------------------------------------------------------------------
__global__ void k_rmul(const float* __restrict__ g, const float* __restrict__ h,
                       float* __restrict__ rh, int N, int Ch, int HW)
{
    int idx = blockIdx.x * blockDim.x + threadIdx.x;
    int total = N * Ch * HW;
    if (idx >= total) return;
    int n = idx / (Ch * HW);
    int r = idx - n * Ch * HW;
    float rg = g[((size_t)n * 2 * Ch + Ch) * HW + r];
    rh[idx] = rg * h[idx];
}

__global__ void k_gru_out(const float* __restrict__ g, const float* __restrict__ c,
                          float* __restrict__ h, int N, int Ch, int HW)
{
    int idx = blockIdx.x * blockDim.x + threadIdx.x;
    int total = N * Ch * HW;
    if (idx >= total) return;
    int n = idx / (Ch * HW);
    int r = idx - n * Ch * HW;
    float z = g[((size_t)n * 2 * Ch) * HW + r];
    h[idx] = (1.0f - z) * h[idx] + z * c[idx];
}

__global__ void k_permute_y(const float* __restrict__ y, float* __restrict__ yt,
                            int Bn, int T, int HW)
{
    int idx = blockIdx.x * blockDim.x + threadIdx.x;
    int total = Bn * T * HW;
    if (idx >= total) return;
    int b = idx / (T * HW);
    int r = idx - b * T * HW;
    int t = r / HW;
    int p = r - t * HW;
    yt[((size_t)t * Bn + b) * HW + p] = y[idx];
}

// ---------------------------------------------------------------------------
// Host-side sequencing
// ---------------------------------------------------------------------------
static inline void launch_conv(hipStream_t s,
                               const float* x1, int C1, const float* x2, int C2,
                               const float* w, const float* b, float* out,
                               int N, int Hin, int Win, int Hout, int Wout, int Cout,
                               int mode, int stride, int act)
{
    int totalPos = N * Hout * Wout;
    dim3 grid((unsigned)((totalPos + 63) / 64), (unsigned)((Cout + 31) / 32), 1);
    if (mode == 0)
        wmma_conv_kernel<0><<<grid, dim3(128), 0, s>>>(x1, C1, x2, C2, w, b, out,
                                                       N, Hin, Win, Hout, Wout, Cout, stride, act);
    else
        wmma_conv_kernel<1><<<grid, dim3(128), 0, s>>>(x1, C1, x2, C2, w, b, out,
                                                       N, Hin, Win, Hout, Wout, Cout, stride, act);
}

extern "C" void kernel_launch(void* const* d_in, const int* in_sizes, int n_in,
                              void* d_out, int out_size, void* d_ws, size_t ws_size,
                              hipStream_t stream)
{
    (void)in_sizes; (void)n_in; (void)out_size; (void)ws_size;
    const float* h1_0 = (const float*)d_in[0];
    const float* h2_0 = (const float*)d_in[1];
    const float* h3_0 = (const float*)d_in[2];
    const float* y    = (const float*)d_in[3];
    const float* ew1  = (const float*)d_in[4];  const float* eb1 = (const float*)d_in[5];
    const float* ew2  = (const float*)d_in[6];  const float* eb2 = (const float*)d_in[7];
    const float* ew3  = (const float*)d_in[8];  const float* eb3 = (const float*)d_in[9];
    const float* g3wg = (const float*)d_in[10]; const float* g3bg = (const float*)d_in[11];
    const float* g3wc = (const float*)d_in[12]; const float* g3bc = (const float*)d_in[13];
    const float* g2wg = (const float*)d_in[14]; const float* g2bg = (const float*)d_in[15];
    const float* g2wc = (const float*)d_in[16]; const float* g2bc = (const float*)d_in[17];
    const float* g1wg = (const float*)d_in[18]; const float* g1bg = (const float*)d_in[19];
    const float* g1wc = (const float*)d_in[20]; const float* g1bc = (const float*)d_in[21];
    const float* s3w  = (const float*)d_in[22]; const float* s3b  = (const float*)d_in[23];
    const float* s2w  = (const float*)d_in[24]; const float* s2b  = (const float*)d_in[25];
    const float* s1w  = (const float*)d_in[26]; const float* s1b  = (const float*)d_in[27];
    const float* hdw  = (const float*)d_in[28]; const float* hdb  = (const float*)d_in[29];
    float* out = (float*)d_out;
    float* ws  = (float*)d_ws;

    const int    Bn = 16, T = 10;
    const size_t HW192 = 36864, HW96 = 9216, HW48 = 2304, HW24 = 576;
    const size_t TB = (size_t)T * Bn;   // 160

    // ---- workspace layout (floats); scratch region reused across phases ----
    size_t off = 0;
    const size_t OFF_E3 = off; off += TB * 96 * HW24;
    const size_t OFF_H3 = off; off += (size_t)Bn * 96 * HW24;
    const size_t OFF_H2 = off; off += (size_t)Bn * 64 * HW48;
    const size_t OFF_H1 = off; off += (size_t)Bn * 16 * HW96;
    const size_t OFF_SCR = off;
    // encoder overlay
    const size_t OFF_YT = OFF_SCR;
    const size_t OFF_E1 = OFF_YT + TB * HW192;
    const size_t OFF_E2 = OFF_E1 + TB * 16 * HW96;
    // per-step overlay (encoder temporaries dead by then)
    size_t so = OFF_SCR;
    const size_t OFF_G3  = so; so += (size_t)Bn * 192 * HW24;
    const size_t OFF_RH3 = so; so += (size_t)Bn *  96 * HW24;
    const size_t OFF_C3  = so; so += (size_t)Bn *  96 * HW24;
    const size_t OFF_O3  = so; so += (size_t)Bn *  64 * HW48;
    const size_t OFF_G2  = so; so += (size_t)Bn * 128 * HW48;
    const size_t OFF_RH2 = so; so += (size_t)Bn *  64 * HW48;
    const size_t OFF_C2  = so; so += (size_t)Bn *  64 * HW48;
    const size_t OFF_O2  = so; so += (size_t)Bn *  16 * HW96;
    const size_t OFF_G1  = so; so += (size_t)Bn *  32 * HW96;
    const size_t OFF_RH1 = so; so += (size_t)Bn *  16 * HW96;
    const size_t OFF_C1  = so; so += (size_t)Bn *  16 * HW96;
    const size_t OFF_O1  = so; so += (size_t)Bn *   8 * HW192;

    // ---- hidden-state init (capture-legal d2d async copies) ----
    hipMemcpyAsync(ws + OFF_H1, h1_0, (size_t)Bn * 16 * HW96 * sizeof(float),
                   hipMemcpyDeviceToDevice, stream);
    hipMemcpyAsync(ws + OFF_H2, h2_0, (size_t)Bn * 64 * HW48 * sizeof(float),
                   hipMemcpyDeviceToDevice, stream);
    hipMemcpyAsync(ws + OFF_H3, h3_0, (size_t)Bn * 96 * HW24 * sizeof(float),
                   hipMemcpyDeviceToDevice, stream);

    // ---- [B,T,H,W] -> [T*B,1,H,W] ----
    {
        int total = Bn * T * (int)HW192;
        k_permute_y<<<(total + 255) / 256, 256, 0, stream>>>(y, ws + OFF_YT, Bn, T, (int)HW192);
    }

    // ---- encoder: three stride-2 3x3 convs with leaky-relu ----
    launch_conv(stream, ws + OFF_YT,  1, nullptr, 0, ew1, eb1, ws + OFF_E1, (int)TB, 192, 192, 96, 96, 16, 0, 2, 1);
    launch_conv(stream, ws + OFF_E1, 16, nullptr, 0, ew2, eb2, ws + OFF_E2, (int)TB,  96,  96, 48, 48, 64, 0, 2, 1);
    launch_conv(stream, ws + OFF_E2, 64, nullptr, 0, ew3, eb3, ws + OFF_E3, (int)TB,  48,  48, 24, 24, 96, 0, 2, 1);

    float* H3 = ws + OFF_H3;
    float* H2 = ws + OFF_H2;
    float* H1 = ws + OFF_H1;

    for (int t = 0; t < T; ++t) {
        const float* x3 = ws + OFF_E3 + (size_t)t * Bn * 96 * HW24;

        // ---- ConvGRU level 3 (24x24, Ch=96) ----
        launch_conv(stream, x3, 96, H3, 96, g3wg, g3bg, ws + OFF_G3, Bn, 24, 24, 24, 24, 192, 0, 1, 2);
        { int n = Bn * 96 * (int)HW24;
          k_rmul<<<(n + 255) / 256, 256, 0, stream>>>(ws + OFF_G3, H3, ws + OFF_RH3, Bn, 96, (int)HW24); }
        launch_conv(stream, x3, 96, ws + OFF_RH3, 96, g3wc, g3bc, ws + OFF_C3, Bn, 24, 24, 24, 24, 96, 0, 1, 1);
        { int n = Bn * 96 * (int)HW24;
          k_gru_out<<<(n + 255) / 256, 256, 0, stream>>>(ws + OFF_G3, ws + OFF_C3, H3, Bn, 96, (int)HW24); }
        launch_conv(stream, H3, 96, nullptr, 0, s3w, s3b, ws + OFF_O3, Bn, 24, 24, 48, 48, 64, 1, 2, 1);

        // ---- ConvGRU level 2 (48x48, Ch=64) ----
        launch_conv(stream, ws + OFF_O3, 64, H2, 64, g2wg, g2bg, ws + OFF_G2, Bn, 48, 48, 48, 48, 128, 0, 1, 2);
        { int n = Bn * 64 * (int)HW48;
          k_rmul<<<(n + 255) / 256, 256, 0, stream>>>(ws + OFF_G2, H2, ws + OFF_RH2, Bn, 64, (int)HW48); }
        launch_conv(stream, ws + OFF_O3, 64, ws + OFF_RH2, 64, g2wc, g2bc, ws + OFF_C2, Bn, 48, 48, 48, 48, 64, 0, 1, 1);
        { int n = Bn * 64 * (int)HW48;
          k_gru_out<<<(n + 255) / 256, 256, 0, stream>>>(ws + OFF_G2, ws + OFF_C2, H2, Bn, 64, (int)HW48); }
        launch_conv(stream, H2, 64, nullptr, 0, s2w, s2b, ws + OFF_O2, Bn, 48, 48, 96, 96, 16, 1, 2, 1);

        // ---- ConvGRU level 1 (96x96, Ch=16) ----
        launch_conv(stream, ws + OFF_O2, 16, H1, 16, g1wg, g1bg, ws + OFF_G1, Bn, 96, 96, 96, 96, 32, 0, 1, 2);
        { int n = Bn * 16 * (int)HW96;
          k_rmul<<<(n + 255) / 256, 256, 0, stream>>>(ws + OFF_G1, H1, ws + OFF_RH1, Bn, 16, (int)HW96); }
        launch_conv(stream, ws + OFF_O2, 16, ws + OFF_RH1, 16, g1wc, g1bc, ws + OFF_C1, Bn, 96, 96, 96, 96, 16, 0, 1, 1);
        { int n = Bn * 16 * (int)HW96;
          k_gru_out<<<(n + 255) / 256, 256, 0, stream>>>(ws + OFF_G1, ws + OFF_C1, H1, Bn, 16, (int)HW96); }
        launch_conv(stream, H1, 16, nullptr, 0, s1w, s1b, ws + OFF_O1, Bn, 96, 96, 192, 192, 8, 1, 2, 1);

        // ---- head 3x3 conv 8->1, write directly into d_out[t] ----
        launch_conv(stream, ws + OFF_O1, 8, nullptr, 0, hdw, hdb,
                    out + (size_t)t * Bn * HW192, Bn, 192, 192, 192, 192, 1, 0, 1, 0);
    }
}
